// TensorNetInteraction_12189117186390
// MI455X (gfx1250) — compile-verified, matrix operand order checked by reading, not run on hardware
//
#include <hip/hip_runtime.h>
#include <hip/hip_bf16.h>

// TensorNet interaction for MI455X (gfx1250), wave32 + WMMA bf16.
// Compact irrep storage: per node, I=1, A=3, S=6 components x 128 channels.
// Weights pre-packed to bf16 in WMMA-fragment-major layout (32B/lane loads).
// Wave-uniform full-tile fast paths avoid per-element EXEC guards.

#define UNITS 128
#define NUM_RBF 64
#define H1C 128
#define H2C 256
#define H3C 384
#define CUTOFF_F 5.0f
#define PI_OVER_CUTOFF 0.6283185307179586f

typedef __attribute__((ext_vector_type(16))) __bf16 v16bf;
typedef __attribute__((ext_vector_type(8)))  float  v8f;

__device__ __forceinline__ float silu_f(float x) { return x / (1.0f + __expf(-x)); }

// K offset (within a 32-wide K window) for half-index h, given lane group.
// lane<16: h=0..7 -> K 0..7, h=8..15 -> K 16..23 ; lane>=16: +8.
__device__ __forceinline__ int kmap(int klo, int h) {
  return klo + ((h & 8) << 1) + (h & 7);
}

// Load an A fragment: 8 contiguous f32 at p, 8 more at p+16 (vectorized b128).
__device__ __forceinline__ v16bf load_afrag(const float* p) {
  float4 q0 = *(const float4*)(p);
  float4 q1 = *(const float4*)(p + 4);
  float4 q2 = *(const float4*)(p + 16);
  float4 q3 = *(const float4*)(p + 20);
  v16bf f;
  f[0]  = (__bf16)q0.x; f[1]  = (__bf16)q0.y; f[2]  = (__bf16)q0.z; f[3]  = (__bf16)q0.w;
  f[4]  = (__bf16)q1.x; f[5]  = (__bf16)q1.y; f[6]  = (__bf16)q1.z; f[7]  = (__bf16)q1.w;
  f[8]  = (__bf16)q2.x; f[9]  = (__bf16)q2.y; f[10] = (__bf16)q2.z; f[11] = (__bf16)q2.w;
  f[12] = (__bf16)q3.x; f[13] = (__bf16)q3.y; f[14] = (__bf16)q3.z; f[15] = (__bf16)q3.w;
  return f;
}

// ---------------------------------------------------------------------------
// Pack a (K x Ncol) f32 weight matrix into bf16 fragment-major layout:
// P[((kt*nn + nt)*32 + lane)*16 + h] = W[(kt*32 + kmap(lane,h)) * Ncol + nt*16 + sub]
// ---------------------------------------------------------------------------
__global__ void __launch_bounds__(256)
pack_w_kernel(const float* __restrict__ W, __bf16* __restrict__ P, int K, int Ncol) {
  const int nn = Ncol >> 4;
  const int total = K * Ncol;
  int idx = blockIdx.x * blockDim.x + threadIdx.x;
  if (idx >= total) return;
  int h    = idx & 15;
  int lane = (idx >> 4) & 31;
  int tile = idx >> 9;
  int nt = tile % nn;
  int kt = tile / nn;
  int sub = lane & 15;
  int klo = (lane >> 4) << 3;
  int k = kt * 32 + kmap(klo, h);
  int c = nt * 16 + sub;
  P[idx] = (__bf16)W[(size_t)k * Ncol + c];
}

// ---------------------------------------------------------------------------
// Node preprocess: normalize X, decompose into compact irreps I0/A0/S0.
// ---------------------------------------------------------------------------
__global__ void __launch_bounds__(256)
node_prep_kernel(const float* __restrict__ X,
                 float* __restrict__ I0, float* __restrict__ A0,
                 float* __restrict__ S0, int N) {
  int idx = blockIdx.x * blockDim.x + threadIdx.x;
  if (idx >= N * UNITS) return;
  int n = idx / UNITS, u = idx - n * UNITS;
  const float* xp = X + (size_t)n * 9 * UNITS + u;
  float t[9];
  float nrm = 1.0f;
#pragma unroll
  for (int c = 0; c < 9; ++c) { float v = xp[(size_t)c * UNITS]; t[c] = v; nrm += v * v; }
  float rn = 1.0f / nrm;
#pragma unroll
  for (int c = 0; c < 9; ++c) t[c] *= rn;
  float tr = (t[0] + t[4] + t[8]) * (1.0f / 3.0f);
  I0[idx] = tr;
  float* ap = A0 + ((size_t)n * 3) * UNITS + u;
  ap[0]         = 0.5f * (t[1] - t[3]);
  ap[UNITS]     = 0.5f * (t[2] - t[6]);
  ap[2 * UNITS] = 0.5f * (t[5] - t[7]);
  float* sp = S0 + ((size_t)n * 6) * UNITS + u;
  sp[0]         = t[0] - tr;
  sp[UNITS]     = 0.5f * (t[1] + t[3]);
  sp[2 * UNITS] = 0.5f * (t[2] + t[6]);
  sp[3 * UNITS] = t[4] - tr;
  sp[4 * UNITS] = 0.5f * (t[5] + t[7]);
  sp[5 * UNITS] = t[8] - tr;
}

// ---------------------------------------------------------------------------
// Generic (M x 128) @ (128 x 128) bf16-WMMA GEMM with packed weights.
// Block=128 thr (4 waves), 16 rows/wave, 8 column tiles in accumulators.
// Wave-uniform fast path for full tiles (no EXEC guards).
// ---------------------------------------------------------------------------
__global__ void __launch_bounds__(128)
gemm_mix_kernel(const float* __restrict__ A, const __bf16* __restrict__ Wp,
                float* __restrict__ Out, int M) {
  const int lane = threadIdx.x & 31;
  const int wv   = threadIdx.x >> 5;
  const int rowbase = blockIdx.x * 64 + wv * 16;
  if (rowbase >= M) return;                 // wave-uniform: EXEC all-ones for WMMA
  const int sub = lane & 15;
  const int klo = (lane >> 4) << 3;
  const bool full = (rowbase + 16 <= M);    // wave-uniform
  const int arow = rowbase + sub;

  v16bf af[4];
  if (full) {
    const float* ap = A + (size_t)arow * UNITS;
#pragma unroll
    for (int kt = 0; kt < 4; ++kt)
      af[kt] = load_afrag(ap + kt * 32 + klo);
  } else {
    const bool aok = (arow < M);
    const float* ap = A + (size_t)(aok ? arow : rowbase) * UNITS;
    const v16bf zf = {};
#pragma unroll
    for (int kt = 0; kt < 4; ++kt)
      af[kt] = aok ? load_afrag(ap + kt * 32 + klo) : zf;
  }

  const v16bf* W = (const v16bf*)Wp;        // [(kt*8 + nt)*32 + lane]
  float* orow = Out + (size_t)(rowbase + klo) * UNITS + sub;
#pragma unroll
  for (int nt = 0; nt < 8; ++nt) {
    v8f acc = {};
#pragma unroll
    for (int kt = 0; kt < 4; ++kt) {
      v16bf bf = W[(kt * 8 + nt) * 32 + lane];
      acc = __builtin_amdgcn_wmma_f32_16x16x32_bf16(false, af[kt], false, bf,
                                                    (short)0, acc, false, false);
    }
    if (full) {
#pragma unroll
      for (int v = 0; v < 8; ++v)
        orow[(size_t)v * UNITS + nt * 16] = acc[v];
    } else {
#pragma unroll
      for (int v = 0; v < 8; ++v) {
        int r = rowbase + klo + v;
        if (r < M) Out[(size_t)r * UNITS + nt * 16 + sub] = acc[v];
      }
    }
  }
}

// ---------------------------------------------------------------------------
// Zero-fill
// ---------------------------------------------------------------------------
__global__ void zero_kernel(float* __restrict__ p, size_t count) {
  size_t i = (size_t)blockIdx.x * blockDim.x + threadIdx.x;
  if (i < count) p[i] = 0.0f;
}

// ---------------------------------------------------------------------------
// Fused edge MLP (64->128->256->384, SiLU, cosine cutoff) + compact-irrep
// gather/scale/scatter. 16 edges per block, 8 waves. Packed bf16 weights.
// ---------------------------------------------------------------------------
__global__ void __launch_bounds__(256)
edge_mlp_scatter_kernel(const float* __restrict__ edge_attr,
                        const float* __restrict__ edge_weight,
                        const int*   __restrict__ edge_index,
                        const __bf16* __restrict__ P0, const float* __restrict__ bs0,
                        const __bf16* __restrict__ P1, const float* __restrict__ bs1,
                        const __bf16* __restrict__ P2, const float* __restrict__ bs2,
                        const float* __restrict__ Ic, const float* __restrict__ Ac,
                        const float* __restrict__ Sc,
                        float* __restrict__ mI, float* __restrict__ mA,
                        float* __restrict__ mS, int E) {
  __shared__ float h1s[16 * H1C];
  __shared__ float h2s[16 * H2C];
  __shared__ float ffs[16 * H3C];
  __shared__ float cenv[16];
  __shared__ int   esrc[16];
  __shared__ int   edst[16];

  const int ebase = blockIdx.x * 16;
  const int tid = threadIdx.x;
  if (tid < 16) {
    int e = ebase + tid;
    if (e < E) {
      float ew = edge_weight[e];
      cenv[tid] = (ew < CUTOFF_F) ? 0.5f * (__cosf(ew * PI_OVER_CUTOFF) + 1.0f) : 0.0f;
      edst[tid] = edge_index[e];       // row 0 = dst
      esrc[tid] = edge_index[E + e];   // row 1 = src
    } else {
      cenv[tid] = 0.0f; esrc[tid] = 0; edst[tid] = 0;
    }
  }

  const int lane = tid & 31;
  const int wv   = tid >> 5;          // 0..7
  const int sub  = lane & 15;
  const int klo  = (lane >> 4) << 3;

  const v16bf* W0 = (const v16bf*)P0;  // nn=8,  nk=2
  const v16bf* W1 = (const v16bf*)P1;  // nn=16, nk=4
  const v16bf* W2 = (const v16bf*)P2;  // nn=24, nk=8

  // ---- layer 1: (16 x 64) @ (64 x 128), wave w owns cols [w*16, w*16+16)
  {
    v16bf af[2];
    if (ebase + 16 <= E) {             // block-uniform fast path
      const float* ap = edge_attr + (size_t)(ebase + sub) * NUM_RBF;
#pragma unroll
      for (int kt = 0; kt < 2; ++kt)
        af[kt] = load_afrag(ap + kt * 32 + klo);
    } else {
      const int arow = ebase + sub;
      const bool aok = arow < E;
      const float* ap = edge_attr + (size_t)(aok ? arow : 0) * NUM_RBF;
      const v16bf zf = {};
#pragma unroll
      for (int kt = 0; kt < 2; ++kt)
        af[kt] = aok ? load_afrag(ap + kt * 32 + klo) : zf;
    }
    const int colg = wv * 16 + sub;
    v8f acc = {};
#pragma unroll
    for (int kt = 0; kt < 2; ++kt) {
      v16bf bf = W0[(kt * 8 + wv) * 32 + lane];
      acc = __builtin_amdgcn_wmma_f32_16x16x32_bf16(false, af[kt], false, bf,
                                                    (short)0, acc, false, false);
    }
    float b = bs0[colg];
#pragma unroll
    for (int v = 0; v < 8; ++v)
      h1s[(klo + v) * H1C + colg] = silu_f(acc[v] + b);
  }
  __syncthreads();

  // ---- layer 2: (16 x 128) @ (128 x 256), wave w owns col tiles {w, w+8}
  {
    v16bf af[4];
#pragma unroll
    for (int kt = 0; kt < 4; ++kt)
      af[kt] = load_afrag(&h1s[sub * H1C + kt * 32 + klo]);
#pragma unroll
    for (int t = 0; t < 2; ++t) {
      const int nt = wv + t * 8;
      const int colg = nt * 16 + sub;
      v8f acc = {};
#pragma unroll
      for (int kt = 0; kt < 4; ++kt) {
        v16bf bf = W1[(kt * 16 + nt) * 32 + lane];
        acc = __builtin_amdgcn_wmma_f32_16x16x32_bf16(false, af[kt], false, bf,
                                                      (short)0, acc, false, false);
      }
      float b = bs1[colg];
#pragma unroll
      for (int v = 0; v < 8; ++v)
        h2s[(klo + v) * H2C + colg] = silu_f(acc[v] + b);
    }
  }
  __syncthreads();

  // ---- layer 3: (16 x 256) @ (256 x 384), wave w owns col tiles {w, w+8, w+16}
  {
    v16bf af[8];
#pragma unroll
    for (int kt = 0; kt < 8; ++kt)
      af[kt] = load_afrag(&h2s[sub * H2C + kt * 32 + klo]);
#pragma unroll
    for (int t = 0; t < 3; ++t) {
      const int nt = wv + t * 8;
      const int colg = nt * 16 + sub;
      v8f acc = {};
#pragma unroll
      for (int kt = 0; kt < 8; ++kt) {
        v16bf bf = W2[(kt * 24 + nt) * 32 + lane];
        acc = __builtin_amdgcn_wmma_f32_16x16x32_bf16(false, af[kt], false, bf,
                                                      (short)0, acc, false, false);
      }
      float b = bs2[colg];
#pragma unroll
      for (int v = 0; v < 8; ++v) {
        int m = klo + v;
        ffs[m * H3C + colg] = silu_f(acc[v] + b) * cenv[m];
      }
    }
  }
  __syncthreads();

  // ---- scatter: msg[dst] += f0*I[src], f1*A[src], f2*S[src] (compact, fp32 atomics)
  for (int e = 0; e < 16; ++e) {
    if (ebase + e >= E) break;
    const int s = esrc[e], d = edst[e];
    const float* fr = ffs + e * H3C;   // fr[3u+k] = f_k(channel u)
    for (int i = tid; i < 1280; i += 256) {
      const int comp = i >> 7;
      const int u = i & 127;
      float fv, xv; float* dp;
      if (comp == 0) {
        fv = fr[u * 3 + 0];
        xv = Ic[(size_t)s * UNITS + u];
        dp = mI + (size_t)d * UNITS + u;
      } else if (comp < 4) {
        int k = comp - 1;
        fv = fr[u * 3 + 1];
        xv = Ac[((size_t)s * 3 + k) * UNITS + u];
        dp = mA + ((size_t)d * 3 + k) * UNITS + u;
      } else {
        int k = comp - 4;
        fv = fr[u * 3 + 2];
        xv = Sc[((size_t)s * 6 + k) * UNITS + u];
        dp = mS + ((size_t)d * 6 + k) * UNITS + u;
      }
      unsafeAtomicAdd(dp, fv * xv);
    }
  }
}

// ---------------------------------------------------------------------------
// Helpers for 3x3 algebra
// ---------------------------------------------------------------------------
__device__ __forceinline__ void rebuild33(float i, const float* a, const float* s, float* t) {
  t[0] =  i + s[0];    t[1] =  a[0] + s[1]; t[2] =  a[1] + s[2];
  t[3] = -a[0] + s[1]; t[4] =  i + s[3];    t[5] =  a[2] + s[4];
  t[6] = -a[1] + s[2]; t[7] = -a[2] + s[4]; t[8] =  i + s[5];
}

__device__ __forceinline__ void mm33(const float* a, const float* b, float* c) {
#pragma unroll
  for (int i = 0; i < 3; ++i)
#pragma unroll
    for (int j = 0; j < 3; ++j)
      c[i * 3 + j] = a[i * 3 + 0] * b[0 * 3 + j] +
                     a[i * 3 + 1] * b[1 * 3 + j] +
                     a[i * 3 + 2] * b[2 * 3 + j];
}

// ---------------------------------------------------------------------------
// Compose: Cm = msg@Y + Y@msg, decompose, divide by (|Cm|^2 + 1), store compact.
// ---------------------------------------------------------------------------
__global__ void __launch_bounds__(256)
compose_kernel(const float* __restrict__ mI, const float* __restrict__ mA,
               const float* __restrict__ mS,
               const float* __restrict__ Ic, const float* __restrict__ Ac,
               const float* __restrict__ Sc,
               float* __restrict__ cI, float* __restrict__ cA,
               float* __restrict__ cS, int N) {
  int idx = blockIdx.x * blockDim.x + threadIdx.x;
  if (idx >= N * UNITS) return;
  int n = idx / UNITS, u = idx - n * UNITS;

  float ya[3], ys[6], ma[3], ms[6];
  float yi = Ic[idx], mi = mI[idx];
#pragma unroll
  for (int k = 0; k < 3; ++k) {
    ya[k] = Ac[((size_t)n * 3 + k) * UNITS + u];
    ma[k] = mA[((size_t)n * 3 + k) * UNITS + u];
  }
#pragma unroll
  for (int k = 0; k < 6; ++k) {
    ys[k] = Sc[((size_t)n * 6 + k) * UNITS + u];
    ms[k] = mS[((size_t)n * 6 + k) * UNITS + u];
  }
  float y[9], m[9], c1[9], c2[9], cm[9];
  rebuild33(yi, ya, ys, y);
  rebuild33(mi, ma, ms, m);
  mm33(m, y, c1);
  mm33(y, m, c2);
  float nrm = 1.0f;
#pragma unroll
  for (int k = 0; k < 9; ++k) { cm[k] = c1[k] + c2[k]; nrm += cm[k] * cm[k]; }
  float rn = 1.0f / nrm;
  float tr = (cm[0] + cm[4] + cm[8]) * (1.0f / 3.0f);
  cI[idx] = tr * rn;
  float* ap = cA + ((size_t)n * 3) * UNITS + u;
  ap[0]         = 0.5f * (cm[1] - cm[3]) * rn;
  ap[UNITS]     = 0.5f * (cm[2] - cm[6]) * rn;
  ap[2 * UNITS] = 0.5f * (cm[5] - cm[7]) * rn;
  float* sp = cS + ((size_t)n * 6) * UNITS + u;
  sp[0]         = (cm[0] - tr) * rn;
  sp[UNITS]     = 0.5f * (cm[1] + cm[3]) * rn;
  sp[2 * UNITS] = 0.5f * (cm[2] + cm[6]) * rn;
  sp[3 * UNITS] = (cm[4] - tr) * rn;
  sp[4 * UNITS] = 0.5f * (cm[5] + cm[7]) * rn;
  sp[5 * UNITS] = (cm[8] - tr) * rn;
}

// ---------------------------------------------------------------------------
// Finalize: out = Xn + dX + dX@dX   (Xn recomputed from input X)
// ---------------------------------------------------------------------------
__global__ void __launch_bounds__(256)
finalize_kernel(const float* __restrict__ X,
                const float* __restrict__ dI, const float* __restrict__ dA,
                const float* __restrict__ dS,
                float* __restrict__ out, int N) {
  int idx = blockIdx.x * blockDim.x + threadIdx.x;
  if (idx >= N * UNITS) return;
  int n = idx / UNITS, u = idx - n * UNITS;

  float da[3], ds[6];
  float di = dI[idx];
#pragma unroll
  for (int k = 0; k < 3; ++k) da[k] = dA[((size_t)n * 3 + k) * UNITS + u];
#pragma unroll
  for (int k = 0; k < 6; ++k) ds[k] = dS[((size_t)n * 6 + k) * UNITS + u];
  float d[9], dd[9];
  rebuild33(di, da, ds, d);
  mm33(d, d, dd);

  const float* xp = X + (size_t)n * 9 * UNITS + u;
  float t[9];
  float nrm = 1.0f;
#pragma unroll
  for (int c = 0; c < 9; ++c) { float v = xp[(size_t)c * UNITS]; t[c] = v; nrm += v * v; }
  float rn = 1.0f / nrm;
  float* op = out + (size_t)n * 9 * UNITS + u;
#pragma unroll
  for (int c = 0; c < 9; ++c) op[(size_t)c * UNITS] = t[c] * rn + d[c] + dd[c];
}

// ---------------------------------------------------------------------------
// Launch
// ---------------------------------------------------------------------------
extern "C" void kernel_launch(void* const* d_in, const int* in_sizes, int n_in,
                              void* d_out, int out_size, void* d_ws, size_t ws_size,
                              hipStream_t stream) {
  const float* X           = (const float*)d_in[0];
  const int*   edge_index  = (const int*)d_in[1];
  const float* edge_weight = (const float*)d_in[2];
  const float* edge_attr   = (const float*)d_in[3];
  const float* ws0 = (const float*)d_in[10];
  const float* bs0 = (const float*)d_in[11];
  const float* ws1 = (const float*)d_in[12];
  const float* bs1 = (const float*)d_in[13];
  const float* ws2 = (const float*)d_in[14];
  const float* bs2 = (const float*)d_in[15];
  const float* wt[6] = { (const float*)d_in[16], (const float*)d_in[17],
                         (const float*)d_in[18], (const float*)d_in[19],
                         (const float*)d_in[20], (const float*)d_in[21] };

  const int N = in_sizes[0] / (9 * UNITS);
  const int E = in_sizes[2];

  const size_t setSz = (size_t)N * 10 * UNITS;   // floats per compact irrep set
  float* buf0 = (float*)d_ws;                    // irreps pre-mix, later Cm-div
  float* buf1 = buf0 + setSz;                    // mixed Y irreps
  float* buf2 = buf1 + setSz;                    // msg accumulators, later dX irreps

  // packed bf16 weights after the float buffers (32B-aligned offsets)
  __bf16* p_ws0 = (__bf16*)(buf2 + setSz);       // 64*128   = 8192
  __bf16* p_ws1 = p_ws0 + 8192;                  // 128*256  = 32768
  __bf16* p_ws2 = p_ws1 + 32768;                 // 256*384  = 98304
  __bf16* p_wt[6];
  p_wt[0] = p_ws2 + 98304;
  for (int i = 1; i < 6; ++i) p_wt[i] = p_wt[i - 1] + 16384;   // 128*128 each

  float* I0 = buf0;  float* A0 = buf0 + (size_t)N * UNITS;  float* S0 = buf0 + (size_t)4 * N * UNITS;
  float* Ic = buf1;  float* Ac = buf1 + (size_t)N * UNITS;  float* Sc = buf1 + (size_t)4 * N * UNITS;
  float* mI = buf2;  float* mA = buf2 + (size_t)N * UNITS;  float* mS = buf2 + (size_t)4 * N * UNITS;
  float* cI = buf0;  float* cA = buf0 + (size_t)N * UNITS;  float* cS = buf0 + (size_t)4 * N * UNITS;
  float* dI = buf2;  float* dA = buf2 + (size_t)N * UNITS;  float* dS = buf2 + (size_t)4 * N * UNITS;

  // --- pack weights to bf16 fragment layout ---
  pack_w_kernel<<<(64 * 128 + 255) / 256, 256, 0, stream>>>(ws0, p_ws0, 64, 128);
  pack_w_kernel<<<(128 * 256 + 255) / 256, 256, 0, stream>>>(ws1, p_ws1, 128, 256);
  pack_w_kernel<<<(256 * 384 + 255) / 256, 256, 0, stream>>>(ws2, p_ws2, 256, 384);
  for (int i = 0; i < 6; ++i)
    pack_w_kernel<<<(128 * 128 + 255) / 256, 256, 0, stream>>>(wt[i], p_wt[i], 128, 128);

  const int nbNC = (N * UNITS + 255) / 256;

  node_prep_kernel<<<nbNC, 256, 0, stream>>>(X, I0, A0, S0, N);

  gemm_mix_kernel<<<(N + 63) / 64, 128, 0, stream>>>(I0, p_wt[0], Ic, N);
  gemm_mix_kernel<<<(3 * N + 63) / 64, 128, 0, stream>>>(A0, p_wt[1], Ac, 3 * N);
  gemm_mix_kernel<<<(6 * N + 63) / 64, 128, 0, stream>>>(S0, p_wt[2], Sc, 6 * N);

  zero_kernel<<<(unsigned)((setSz + 255) / 256), 256, 0, stream>>>(buf2, setSz);

  edge_mlp_scatter_kernel<<<(E + 15) / 16, 256, 0, stream>>>(
      edge_attr, edge_weight, edge_index,
      p_ws0, bs0, p_ws1, bs1, p_ws2, bs2,
      Ic, Ac, Sc, mI, mA, mS, E);

  compose_kernel<<<nbNC, 256, 0, stream>>>(mI, mA, mS, Ic, Ac, Sc, cI, cA, cS, N);

  gemm_mix_kernel<<<(N + 63) / 64, 128, 0, stream>>>(cI, p_wt[3], dI, N);
  gemm_mix_kernel<<<(3 * N + 63) / 64, 128, 0, stream>>>(cA, p_wt[4], dA, 3 * N);
  gemm_mix_kernel<<<(6 * N + 63) / 64, 128, 0, stream>>>(cS, p_wt[5], dS, 6 * N);

  finalize_kernel<<<nbNC, 256, 0, stream>>>(X, dI, dA, dS, (float*)d_out, N);
}
